// Model_50646254355231
// MI455X (gfx1250) — compile-verified, matrix operand order checked by reading, not run on hardware
//
#include <hip/hip_runtime.h>
#include <hip/hip_bf16.h>

#define NUc 20000
#define NIc 20000
#define NEc 20000
#define Dc 128
#define Hc 256
#define Oc 64
#define EFc 128
#define EUIc 640000
#define INCc 320000
#define NPAIRc 100000
#define NUNIc (NUc + NIc)

#define CDIV(a,b) (((a)+(b)-1)/(b))

typedef __attribute__((ext_vector_type(16))) __bf16       bf16x16;
typedef __attribute__((ext_vector_type(8)))  float        v8f;
typedef __attribute__((ext_vector_type(8)))  unsigned int u32x8;

// ---------- helpers ----------
__device__ __forceinline__ __bf16 f2bf(float f) {
    unsigned int u = __float_as_uint(f);
    u += 0x7FFFu + ((u >> 16) & 1u);            // round-to-nearest-even
    unsigned short s = (unsigned short)(u >> 16);
    return __builtin_bit_cast(__bf16, s);
}

__device__ __forceinline__ void atomicMaxF(float* addr, float val) {
    if (val >= 0.0f) atomicMax((int*)addr, __float_as_int(val));
    else             atomicMin((unsigned int*)addr, __float_as_uint(val));
}

// ---------- elementwise / init ----------
__global__ void k_fill(float* p, float v, int n) {
    for (int i = blockIdx.x * blockDim.x + threadIdx.x; i < n; i += gridDim.x * blockDim.x) p[i] = v;
}
__global__ void k_add(float* dst, const float* __restrict__ src, int n) {
    for (int i = blockIdx.x * blockDim.x + threadIdx.x; i < n; i += gridDim.x * blockDim.x) dst[i] += src[i];
}
__global__ void k_scale_copy(float* dst, const float* __restrict__ src, float s, int n) {
    for (int i = blockIdx.x * blockDim.x + threadIdx.x; i < n; i += gridDim.x * blockDim.x) dst[i] = src[i] * s;
}
__global__ void k_count(const int* __restrict__ idx, int n, float* deg, int off) {
    int i = blockIdx.x * blockDim.x + threadIdx.x;
    if (i < n) atomicAdd(&deg[idx[i] + off], 1.0f);
}
__global__ void k_rsqrt_clamp(float* d, int n) {
    int i = blockIdx.x * blockDim.x + threadIdx.x;
    if (i < n) d[i] = rsqrtf(fmaxf(d[i], 1.0f));
}
__global__ void k_recip_clamp(float* d, int n) {
    int i = blockIdx.x * blockDim.x + threadIdx.x;
    if (i < n) d[i] = 1.0f / fmaxf(d[i], 1.0f);
}
__global__ void k_row_scale(float* m, const float* __restrict__ s, int rows, int cols) {
    int n = rows * cols;
    for (int i = blockIdx.x * blockDim.x + threadIdx.x; i < n; i += gridDim.x * blockDim.x)
        m[i] *= s[i / cols];
}
__global__ void k_row_scale_copy(float* dst, const float* __restrict__ src, const float* __restrict__ s,
                                 int rows, int cols) {
    int n = rows * cols;
    for (int i = blockIdx.x * blockDim.x + threadIdx.x; i < n; i += gridDim.x * blockDim.x)
        dst[i] = src[i] * s[i / cols];
}
__global__ void k_unigin(float* hl, const float* __restrict__ xh, const float* __restrict__ tmp,
                         const float* __restrict__ epsp, int n) {
    float c = 1.0f + epsp[0];
    for (int i = blockIdx.x * blockDim.x + threadIdx.x; i < n; i += gridDim.x * blockDim.x)
        hl[i] = 3.0f * fmaxf(c * xh[i] + tmp[i], 0.0f);
}
__global__ void k_elu(float* dst, const float* __restrict__ src, int n) {
    for (int i = blockIdx.x * blockDim.x + threadIdx.x; i < n; i += gridDim.x * blockDim.x) {
        float x = src[i];
        dst[i] = x > 0.0f ? x : (__expf(x) - 1.0f);
    }
}

// ---------- graph scatter kernels (wave per edge/incidence) ----------
__global__ void k_bigcn(const int* __restrict__ uu, const int* __restrict__ ii,
                        const float* __restrict__ X, const float* __restrict__ dinv,
                        float* Xn, int nE) {
    int w = (blockIdx.x * blockDim.x + threadIdx.x) >> 5;
    int lane = threadIdx.x & 31;
    if (w >= nE) return;
    int u = uu[w];
    int v = ii[w] + NUc;
    float wt = dinv[u] * dinv[v];
    float4 a = *(const float4*)(X + (size_t)u * Dc + lane * 4);
    float4 b = *(const float4*)(X + (size_t)v * Dc + lane * 4);
    float* ov = Xn + (size_t)v * Dc + lane * 4;
    float* ou = Xn + (size_t)u * Dc + lane * 4;
    atomicAdd(ov + 0, a.x * wt); atomicAdd(ov + 1, a.y * wt);
    atomicAdd(ov + 2, a.z * wt); atomicAdd(ov + 3, a.w * wt);
    atomicAdd(ou + 0, b.x * wt); atomicAdd(ou + 1, b.y * wt);
    atomicAdd(ou + 2, b.z * wt); atomicAdd(ou + 3, b.w * wt);
}

__global__ void k_scatter_add(const float* __restrict__ src, const int* __restrict__ si,
                              const int* __restrict__ di, float* dst, int n, int cols) {
    int w = (blockIdx.x * blockDim.x + threadIdx.x) >> 5;
    int lane = threadIdx.x & 31;
    if (w >= n) return;
    int sr = si[w], dr = di[w];
    for (int c = lane * 4; c < cols; c += 128) {
        float4 v = *(const float4*)(src + (size_t)sr * cols + c);
        float* dp = dst + (size_t)dr * cols + c;
        atomicAdd(dp + 0, v.x); atomicAdd(dp + 1, v.y);
        atomicAdd(dp + 2, v.z); atomicAdd(dp + 3, v.w);
    }
}

__global__ void k_rowdot(const float* __restrict__ Y, const float* __restrict__ a,
                         float* out, int rows, int cols) {
    int w = (blockIdx.x * blockDim.x + threadIdx.x) >> 5;
    int lane = threadIdx.x & 31;
    if (w >= rows) return;
    float p = 0.0f;
    for (int c = lane; c < cols; c += 32) p += Y[(size_t)w * cols + c] * a[c];
    for (int o = 16; o > 0; o >>= 1) p += __shfl_xor(p, o, 32);
    if (lane == 0) out[w] = p;
}

__global__ void k_score_max(const float* __restrict__ edot, const int* __restrict__ eidx,
                            const int* __restrict__ vidx, float* score, float* mv, int n) {
    int i = blockIdx.x * blockDim.x + threadIdx.x;
    if (i >= n) return;
    float s = edot[eidx[i]];
    s = s > 0.0f ? s : 0.2f * s;       // leaky_relu, GAT slope
    score[i] = s;
    atomicMaxF(&mv[vidx[i]], s);
}

__global__ void k_exp_den(float* score, const float* __restrict__ mv,
                          const int* __restrict__ vidx, float* den, int n) {
    int i = blockIdx.x * blockDim.x + threadIdx.x;
    if (i >= n) return;
    int v = vidx[i];
    float e = __expf(score[i] - mv[v]);
    score[i] = e;
    atomicAdd(&den[v], e);
}

__global__ void k_gat_scatter(const float* __restrict__ Y, const int* __restrict__ eidx,
                              const int* __restrict__ vidx, const float* __restrict__ ex,
                              const float* __restrict__ den, float* out, int n, int cols) {
    int w = (blockIdx.x * blockDim.x + threadIdx.x) >> 5;
    int lane = threadIdx.x & 31;
    if (w >= n) return;
    int e = eidx[w], v = vidx[w];
    float wt = ex[w] / fmaxf(den[v], 1e-12f);
    for (int c = lane * 4; c < cols; c += 128) {
        float4 y = *(const float4*)(Y + (size_t)e * cols + c);
        float* dp = out + (size_t)v * cols + c;
        atomicAdd(dp + 0, y.x * wt); atomicAdd(dp + 1, y.y * wt);
        atomicAdd(dp + 2, y.z * wt); atomicAdd(dp + 3, y.w * wt);
    }
}

// ---------- bf16 staging ----------
__global__ void k_f2bf(const float* __restrict__ src, __bf16* dst, int n) {
    for (int i = blockIdx.x * blockDim.x + threadIdx.x; i < n; i += gridDim.x * blockDim.x)
        dst[i] = f2bf(src[i]);
}
__global__ void k_w2bf_t(const float* __restrict__ W, __bf16* Wt, int K, int N) {
    int n = K * N;
    for (int i = blockIdx.x * blockDim.x + threadIdx.x; i < n; i += gridDim.x * blockDim.x) {
        int k = i / N, c = i % N;
        Wt[(size_t)c * K + k] = f2bf(W[i]);
    }
}
__global__ void k_concat_bf(const float* __restrict__ A, const float* __restrict__ B,
                            __bf16* dst, int rows, int cols) {
    int n = rows * cols * 2;
    for (int i = blockIdx.x * blockDim.x + threadIdx.x; i < n; i += gridDim.x * blockDim.x) {
        int r = i / (2 * cols), c = i % (2 * cols);
        float v = (c < cols) ? A[(size_t)r * cols + c] : B[(size_t)r * cols + (c - cols)];
        dst[i] = f2bf(v);
    }
}

// ---------- WMMA GEMM: C = act(A[M,K]bf16 @ B(Kt stored [N,K])bf16 + bias [+C]) ----------
// Fragment layout per CDNA5 ISA 7.12.2 (16-bit A 16x32): lane<16 -> K {0..7,16..23},
// lane>=16 -> K {8..15,24..31}; pairs packed 2/bf16 per dword -> contiguous dword loads.
__global__ void k_gemm_wmma(const __bf16* __restrict__ A, const __bf16* __restrict__ Bt,
                            const float* __restrict__ bias, float* C,
                            int M, int N, int K, int act, int addC) {
    int lane = threadIdx.x & 31;
    int wid  = threadIdx.x >> 5;
    int tile = blockIdx.x * 4 + wid;
    int tilesN = N >> 4, tilesM = M >> 4;
    if (tile >= tilesM * tilesN) return;
    int tm = tile / tilesN, tn = tile % tilesN;
    int hi = lane >> 4, l15 = lane & 15;

    const unsigned int* pa = (const unsigned int*)(A  + (size_t)(tm * 16 + l15) * K);
    const unsigned int* pb = (const unsigned int*)(Bt + (size_t)(tn * 16 + l15) * K);

    v8f acc = {};
    for (int k0 = 0; k0 < K; k0 += 32) {
        int base = (k0 >> 1) + hi * 4;
        u32x8 ua, ub;
#pragma unroll
        for (int i = 0; i < 4; i++) {
            ua[i]     = pa[base + i];
            ua[i + 4] = pa[base + 8 + i];
            ub[i]     = pb[base + i];
            ub[i + 4] = pb[base + 8 + i];
        }
        bf16x16 af = __builtin_bit_cast(bf16x16, ua);
        bf16x16 bfv = __builtin_bit_cast(bf16x16, ub);
        acc = __builtin_amdgcn_wmma_f32_16x16x32_bf16(false, af, false, bfv,
                                                      (short)0, acc, false, false);
    }
#pragma unroll
    for (int i = 0; i < 8; i++) {
        int row = tm * 16 + i + hi * 8;
        int col = tn * 16 + l15;
        size_t idx = (size_t)row * N + col;
        float o = acc[i] + bias[col];
        if (addC) o += C[idx];
        if (act == 1)      o = fmaxf(o, 0.0f);
        else if (act == 2) o = o > 0.0f ? o : 0.01f * o;   // MLP leaky slope
        C[idx] = o;
    }
}

// ---------- output ----------
__global__ void k_gather_rows(const float* __restrict__ src, const int* __restrict__ idx,
                              float* dst, int npair, int cols) {
    int n = npair * cols;
    for (int i = blockIdx.x * blockDim.x + threadIdx.x; i < n; i += gridDim.x * blockDim.x) {
        int p = i / cols, c = i % cols;
        dst[i] = src[(size_t)idx[p] * cols + c];
    }
}
__global__ void k_scores(float* out, int npair) {
    int w = (blockIdx.x * blockDim.x + threadIdx.x) >> 5;
    int lane = threadIdx.x & 31;
    if (w >= npair) return;
    const float* a = out + (size_t)w * 64;
    const float* b = out + (size_t)npair * 64 + (size_t)w * 64;
    float p = a[lane] * b[lane] + a[lane + 32] * b[lane + 32];
    for (int o = 16; o > 0; o >>= 1) p += __shfl_xor(p, o, 32);
    if (lane == 0) out[(size_t)2 * npair * 64 + w] = p;
}

extern "C" void kernel_launch(void* const* d_in, const int* in_sizes, int n_in,
                              void* d_out_, int out_size, void* d_ws, size_t ws_size,
                              hipStream_t stream) {
    const float* u_emb = (const float*)d_in[0];
    const float* i_emb = (const float*)d_in[1];
    const float* he_emb = (const float*)d_in[2];
    const float* We = (const float*)d_in[3];
    const float* be = (const float*)d_in[4];
    const float* Wc = (const float*)d_in[5];
    const float* bc = (const float*)d_in[6];
    const float* eps = (const float*)d_in[7];
    const float* Wor = (const float*)d_in[8];
    const float* bor = (const float*)d_in[9];
    const float* aor = (const float*)d_in[10];
    const float* Wee = (const float*)d_in[11];
    const float* bee = (const float*)d_in[12];
    const float* aee = (const float*)d_in[13];
    const float* W1 = (const float*)d_in[14];
    const float* b1 = (const float*)d_in[15];
    const float* W2 = (const float*)d_in[16];
    const float* b2 = (const float*)d_in[17];
    const float* W3 = (const float*)d_in[18];
    const float* b3 = (const float*)d_in[19];
    const float* Wt = (const float*)d_in[20];
    const float* bt = (const float*)d_in[21];
    const int* ui_u = (const int*)d_in[23];
    const int* ui_i = (const int*)d_in[24];
    const int* sv = (const int*)d_in[25];
    const int* se = (const int*)d_in[26];
    const int* orv = (const int*)d_in[27];
    const int* ore = (const int*)d_in[28];
    const int* eev = (const int*)d_in[29];
    const int* eee = (const int*)d_in[30];
    const int* or_x = (const int*)d_in[31];
    const int* ee_x = (const int*)d_in[32];
    float* out = (float*)d_out_;

    // workspace bump allocator (256B aligned)
    size_t off = 0;
    char* base = (char*)d_ws;
    auto alloc = [&](size_t bytes) -> char* {
        char* p = base + off;
        off += (bytes + 255) & ~(size_t)255;
        return p;
    };
    float* X0    = (float*)alloc((size_t)NUNIc * Dc * 4);
    float* X1    = (float*)alloc((size_t)NUNIc * Dc * 4);
    float* ACC   = (float*)alloc((size_t)NUNIc * Dc * 4);
    float* DINV  = (float*)alloc((size_t)NUNIc * 4);
    float* U     = (float*)alloc((size_t)NUc * Dc * 4);
    float* EDGEX = (float*)alloc((size_t)NUc * Dc * 4);
    float* XV    = (float*)alloc((size_t)NUc * Dc * 4);
    float* YE128 = (float*)alloc((size_t)NEc * Dc * 4);
    float* DVS   = (float*)alloc((size_t)NUc * 4);
    float* DES   = (float*)alloc((size_t)NEc * 4);
    float* DEOR  = (float*)alloc((size_t)NEc * 4);
    float* DEEE  = (float*)alloc((size_t)NEc * 4);
    float* XH    = (float*)alloc((size_t)NUc * Hc * 4);
    float* YE256 = (float*)alloc((size_t)NEc * Hc * 4);
    float* TMP   = (float*)alloc((size_t)NUc * Hc * 4);
    float* HL    = (float*)alloc((size_t)NUc * Hc * 4);
    float* ORG   = (float*)alloc((size_t)NUc * Hc * 4);
    float* EEG   = (float*)alloc((size_t)NUc * Hc * 4);
    float* EDOT  = (float*)alloc((size_t)NEc * 4);
    float* SC    = (float*)alloc((size_t)INCc * 4);
    float* MV    = (float*)alloc((size_t)NUc * 4);
    float* DEN   = (float*)alloc((size_t)NUc * 4);
    float* T1    = (float*)alloc((size_t)NUc * Hc * 4);
    float* T2    = (float*)alloc((size_t)NUc * Hc * 4);
    float* T3    = (float*)alloc((size_t)NUc * Oc * 4);
    float* HX    = (float*)alloc((size_t)NUc * Oc * 4);
    __bf16* ABF1 = (__bf16*)alloc((size_t)NUc * 512 * 2);  // edgex-bf / u-bf / concat-bf
    __bf16* ABF2 = (__bf16*)alloc((size_t)NUc * Hc * 2);   // hl-bf / t1-bf / t2-bf
    __bf16* WBF  = (__bf16*)alloc((size_t)512 * 256 * 2);  // transposed weight staging

    auto zero = [&](float* p, int n) { k_fill<<<CDIV(n, 256), 256, 0, stream>>>(p, 0.0f, n); };
    auto gemm = [&](const __bf16* A, const __bf16* Bt, const float* bias, float* C,
                    int M, int N, int K, int act, int addC) {
        int tiles = (M >> 4) * (N >> 4);
        k_gemm_wmma<<<CDIV(tiles, 4), 128, 0, stream>>>(A, Bt, bias, C, M, N, K, act, addC);
    };
    auto w2bf = [&](const float* W, int K, int N) {
        k_w2bf_t<<<CDIV(K * N, 256), 256, 0, stream>>>(W, WBF, K, N);
    };

    // ===== degrees =====
    zero(DINV, NUNIc);
    k_count<<<CDIV(EUIc, 256), 256, 0, stream>>>(ui_u, EUIc, DINV, 0);
    k_count<<<CDIV(EUIc, 256), 256, 0, stream>>>(ui_i, EUIc, DINV, NUc);
    k_rsqrt_clamp<<<CDIV(NUNIc, 256), 256, 0, stream>>>(DINV, NUNIc);
    zero(DVS, NUc);
    k_count<<<CDIV(INCc, 256), 256, 0, stream>>>(sv, INCc, DVS, 0);
    k_rsqrt_clamp<<<CDIV(NUc, 256), 256, 0, stream>>>(DVS, NUc);
    zero(DES, NEc);
    k_count<<<CDIV(INCc, 256), 256, 0, stream>>>(se, INCc, DES, 0);
    k_recip_clamp<<<CDIV(NEc, 256), 256, 0, stream>>>(DES, NEc);
    zero(DEOR, NEc);
    k_count<<<CDIV(INCc, 256), 256, 0, stream>>>(ore, INCc, DEOR, 0);
    k_recip_clamp<<<CDIV(NEc, 256), 256, 0, stream>>>(DEOR, NEc);
    zero(DEEE, NEc);
    k_count<<<CDIV(INCc, 256), 256, 0, stream>>>(eee, INCc, DEEE, 0);
    k_recip_clamp<<<CDIV(NEc, 256), 256, 0, stream>>>(DEEE, NEc);

    // ===== bipartite LightGCN: 3 smoothings, acc = sum of hops =====
    hipMemcpyAsync(X0, u_emb, (size_t)NUc * Dc * 4, hipMemcpyDeviceToDevice, stream);
    hipMemcpyAsync(X0 + (size_t)NUc * Dc, i_emb, (size_t)NIc * Dc * 4, hipMemcpyDeviceToDevice, stream);
    hipMemcpyAsync(ACC, X0, (size_t)NUNIc * Dc * 4, hipMemcpyDeviceToDevice, stream);
    float* Xc = X0; float* Xn = X1;
    for (int l = 0; l < 3; l++) {
        zero(Xn, NUNIc * Dc);
        k_bigcn<<<CDIV(EUIc, 8), 256, 0, stream>>>(ui_u, ui_i, Xc, DINV, Xn, EUIc);
        k_add<<<CDIV(NUNIc * Dc, 256), 256, 0, stream>>>(ACC, Xn, NUNIc * Dc);
        float* t = Xc; Xc = Xn; Xn = t;
    }
    k_scale_copy<<<CDIV(NUc * Dc, 256), 256, 0, stream>>>(U, ACC, 0.25f, NUc * Dc);

    // ===== hgnn_smooth(he_emb) -> edge_x ; u += edge_x @ We + be =====
    k_row_scale_copy<<<CDIV(NUc * EFc, 256), 256, 0, stream>>>(XV, he_emb, DVS, NUc, EFc);
    zero(YE128, NEc * EFc);
    k_scatter_add<<<CDIV(INCc, 8), 256, 0, stream>>>(XV, sv, se, YE128, INCc, EFc);
    k_row_scale<<<CDIV(NEc * EFc, 256), 256, 0, stream>>>(YE128, DES, NEc, EFc);
    zero(EDGEX, NUc * EFc);
    k_scatter_add<<<CDIV(INCc, 8), 256, 0, stream>>>(YE128, se, sv, EDGEX, INCc, EFc);
    k_row_scale<<<CDIV(NUc * EFc, 256), 256, 0, stream>>>(EDGEX, DVS, NUc, EFc);
    k_f2bf<<<CDIV(NUc * EFc, 256), 256, 0, stream>>>(EDGEX, ABF1, NUc * EFc);
    w2bf(We, EFc, Dc);
    gemm(ABF1, WBF, be, U, NUc, Dc, EFc, 0, 1);   // U += edge_x @ We + be

    // u in bf16 (reused by 3 GEMMs)
    k_f2bf<<<CDIV(NUc * Dc, 256), 256, 0, stream>>>(U, ABF1, NUc * Dc);

    // ===== unigin -> hyper_lats = 3 * relu((1+eps)*Xh + e2v(v2e_mean(Xh))) =====
    w2bf(Wc, Dc, Hc);
    gemm(ABF1, WBF, bc, XH, NUc, Hc, Dc, 0, 0);
    zero(YE256, NEc * Hc);
    k_scatter_add<<<CDIV(INCc, 8), 256, 0, stream>>>(XH, sv, se, YE256, INCc, Hc);
    k_row_scale<<<CDIV(NEc * Hc, 256), 256, 0, stream>>>(YE256, DES, NEc, Hc);
    zero(TMP, NUc * Hc);
    k_scatter_add<<<CDIV(INCc, 8), 256, 0, stream>>>(YE256, se, sv, TMP, INCc, Hc);
    k_unigin<<<CDIV(NUc * Hc, 256), 256, 0, stream>>>(HL, XH, TMP, eps, NUc * Hc);

    // ===== unigat (or / ee) =====
    struct GatCfg { const float* W; const float* b; const float* a;
                    const int* v; const int* e; const float* de; float* dst; };
    GatCfg cfgs[2] = { { Wor, bor, aor, orv, ore, DEOR, ORG },
                       { Wee, bee, aee, eev, eee, DEEE, EEG } };
    for (int g = 0; g < 2; g++) {
        GatCfg c = cfgs[g];
        w2bf(c.W, Dc, Hc);
        gemm(ABF1, WBF, c.b, XH, NUc, Hc, Dc, 0, 0);          // theta(u) (XH reused)
        zero(YE256, NEc * Hc);
        k_scatter_add<<<CDIV(INCc, 8), 256, 0, stream>>>(XH, c.v, c.e, YE256, INCc, Hc);
        k_row_scale<<<CDIV(NEc * Hc, 256), 256, 0, stream>>>(YE256, c.de, NEc, Hc);
        k_rowdot<<<CDIV(NEc, 8), 256, 0, stream>>>(YE256, c.a, EDOT, NEc, Hc);
        k_fill<<<CDIV(NUc, 256), 256, 0, stream>>>(MV, -3.4e38f, NUc);
        k_score_max<<<CDIV(INCc, 256), 256, 0, stream>>>(EDOT, c.e, c.v, SC, MV, INCc);
        zero(DEN, NUc);
        k_exp_den<<<CDIV(INCc, 256), 256, 0, stream>>>(SC, MV, c.v, DEN, INCc);
        zero(TMP, NUc * Hc);
        k_gat_scatter<<<CDIV(INCc, 8), 256, 0, stream>>>(YE256, c.e, c.v, SC, DEN, TMP, INCc, Hc);
        k_elu<<<CDIV(NUc * Hc, 256), 256, 0, stream>>>(c.dst, TMP, NUc * Hc);
    }

    // ===== hx = relu(HL @ Wt + bt) =====
    k_f2bf<<<CDIV(NUc * Hc, 256), 256, 0, stream>>>(HL, ABF2, NUc * Hc);
    w2bf(Wt, Hc, Oc);
    gemm(ABF2, WBF, bt, HX, NUc, Oc, Hc, 1, 0);

    // ===== MLP(concat(gat, HL)) + HX, gathered to pairs =====
    for (int g = 0; g < 2; g++) {
        const float* gat = (g == 0) ? ORG : EEG;
        const int* px = (g == 0) ? or_x : ee_x;
        float* dst = out + (size_t)g * NPAIRc * Oc;
        k_concat_bf<<<CDIV(NUc * 512, 256), 256, 0, stream>>>(gat, HL, ABF1, NUc, Hc);
        w2bf(W1, 2 * Hc, Hc);
        gemm(ABF1, WBF, b1, T1, NUc, Hc, 2 * Hc, 2, 0);
        k_f2bf<<<CDIV(NUc * Hc, 256), 256, 0, stream>>>(T1, ABF2, NUc * Hc);
        w2bf(W2, Hc, Hc);
        gemm(ABF2, WBF, b2, T2, NUc, Hc, Hc, 2, 0);
        k_f2bf<<<CDIV(NUc * Hc, 256), 256, 0, stream>>>(T2, ABF2, NUc * Hc);
        w2bf(W3, Hc, Oc);
        gemm(ABF2, WBF, b3, T3, NUc, Oc, Hc, 0, 0);
        k_add<<<CDIV(NUc * Oc, 256), 256, 0, stream>>>(T3, HX, NUc * Oc);
        k_gather_rows<<<CDIV(NPAIRc * Oc, 256), 256, 0, stream>>>(T3, px, dst, NPAIRc, Oc);
    }

    // ===== scores = rowwise dot(trustor, trustee) =====
    k_scores<<<CDIV(NPAIRc, 8), 256, 0, stream>>>(out, NPAIRc);
}
